// Sidechain_3332894622683
// MI455X (gfx1250) — matrix operand order repaired and checked by reading, not compiled, not run on hardware
//
#include <hip/hip_runtime.h>
#include <hip/hip_bf16.h>
#include <math.h>

// ---------------------------------------------------------------------------
// out[b,l,c] = sum_k x[b,l+k,c] * tanh( sum_c' x[b,l+6,c'] * W[c,c',k] )
// B=8, L=4096, C=128, K=7, Lp=4090
//
// GEMM via V_WMMA_F32_16X16X32_BF16 with fp32 -> bf16 hi/lo split
// (Ah*Bh + Ah*Bl + Al*Bh) for near-fp32 accuracy at bf16-WMMA rates.
// tanh via hardware V_TANH_F32 when available; x tile staged to LDS with
// GLOBAL_LOAD_ASYNC_TO_LDS_B128 (ASYNCcnt) when available.
// ---------------------------------------------------------------------------

typedef __attribute__((ext_vector_type(16))) __bf16 v16bf;
typedef __attribute__((ext_vector_type(8)))  float  v8f;
typedef __attribute__((ext_vector_type(4)))  int    v4i;

#define B_   8
#define L_   4096
#define C_   128
#define K_   7
#define LP_  (L_ - K_ + 1)          // 4090
#define TILES_ ((LP_ + 15) / 16)    // 256 row tiles per batch
#define NFRAG_ELEMS (K_ * 4 * 8 * 32 * 16)   // 114688 = 7*128*128

#if defined(__has_builtin)
#if __has_builtin(__builtin_amdgcn_tanhf)
#define FAST_TANH(x) __builtin_amdgcn_tanhf(x)
#endif
#if __has_builtin(__builtin_amdgcn_global_load_async_to_lds_b128)
#define HAVE_ASYNC_LDS 1
#endif
#endif
#ifndef FAST_TANH
#define FAST_TANH(x) tanhf(x)
#endif

#define AS_GLOBAL __attribute__((address_space(1)))
#define AS_LOCAL  __attribute__((address_space(3)))

// ---------------------------------------------------------------------------
// Prep: split weights into bf16 hi/lo, laid out as ready-to-load WMMA B
// fragments: index = ((((kk*4 + s)*8 + wtile)*32) + lane)*16 + j
// B fragment k order chosen here (contiguous, same as A; 32x16, K x N):
//   lane l: n = l%16, K = (l/16)*16 + j   (j = element 0..15 in v16bf)
// B[K=cin][n=cout] = weights[cout][cin][kk], cin = s*32 + (l/16)*16 + j
// ---------------------------------------------------------------------------
__global__ void prep_weights(const float* __restrict__ w,
                             __bf16* __restrict__ wh,
                             __bf16* __restrict__ wl) {
    int i = blockIdx.x * blockDim.x + threadIdx.x;
    if (i >= NFRAG_ELEMS) return;
    int j     = i & 15;
    int lane  = (i >> 4) & 31;
    int wtile = (i >> 9) & 7;
    int s     = (i >> 12) & 3;
    int kk    = i >> 14;            // 0..6
    int cout  = wtile * 16 + (lane & 15);
    int cin   = s * 32 + (lane >> 4) * 16 + j;
    float f   = w[cout * (C_ * K_) + cin * K_ + kk];
    __bf16 hi = (__bf16)f;
    __bf16 lo = (__bf16)(f - (float)hi);
    wh[i] = hi;
    wl[i] = lo;
}

// ---------------------------------------------------------------------------
// Main kernel: one block = 16 output rows x all 128 channels for one batch.
// 8 waves; wave wtile handles channels [16*wtile, 16*wtile+16).
// ---------------------------------------------------------------------------
__global__ __launch_bounds__(256) void sidechain_kernel(
    const float*  __restrict__ x,
    const __bf16* __restrict__ wh,
    const __bf16* __restrict__ wl,
    float*        __restrict__ out) {

    __shared__ float lds_x[22 * C_];          // rows l0 .. l0+21 of x[b]

    const int bidx  = blockIdx.x / TILES_;
    const int tile  = blockIdx.x % TILES_;
    const int l0    = tile * 16;
    const int tid   = threadIdx.x;
    const int lane  = tid & 31;
    const int wtile = tid >> 5;

    // ---- stage 22 rows of x into LDS (contiguous in memory), zero OOB ----
    {
        const float4* src = (const float4*)(x + ((size_t)bidx * L_ + l0) * C_);
#ifdef HAVE_ASYNC_LDS
        for (int i = tid; i < 22 * (C_ / 4); i += 256) {
            int row = l0 + (i >> 5);          // 32 float4 per row
            if (row < L_) {
                __builtin_amdgcn_global_load_async_to_lds_b128(
                    (AS_GLOBAL v4i*)(src + i),
                    (AS_LOCAL  v4i*)&lds_x[i * 4],
                    0, 0);
            } else {
                *(float4*)&lds_x[i * 4] = make_float4(0.f, 0.f, 0.f, 0.f);
            }
        }
        asm volatile("s_wait_asynccnt 0x0" ::: "memory");
#else
        for (int i = tid; i < 22 * (C_ / 4); i += 256) {
            int row = l0 + (i >> 5);
            float4 v = make_float4(0.f, 0.f, 0.f, 0.f);
            if (row < L_) v = src[i];
            *(float4*)&lds_x[i * 4] = v;
        }
#endif
    }
    __syncthreads();

    const int half = lane >> 4;
    const int am   = lane & 15;               // A-matrix row for this lane

    v8f acc[K_];
    #pragma unroll
    for (int kk = 0; kk < K_; ++kk) acc[kk] = (v8f){};

    // A row m = x[b, l0+6+m, :]  (GEMM driving row is l+6)
    const float* xrow = &lds_x[(am + 6) * C_];

    for (int s = 0; s < 4; ++s) {             // K-steps of 32 over cin
        // A fragment layout (16x32 bf16): lane am, element j:
        //   K = s*32 + half*16 + j
        const int c0 = s * 32 + half * 16;
        float fa[16];
        {
            float4 f0 = *(const float4*)&xrow[c0];
            float4 f1 = *(const float4*)&xrow[c0 + 4];
            float4 f2 = *(const float4*)&xrow[c0 + 8];
            float4 f3 = *(const float4*)&xrow[c0 + 12];
            fa[0]=f0.x;  fa[1]=f0.y;  fa[2]=f0.z;  fa[3]=f0.w;
            fa[4]=f1.x;  fa[5]=f1.y;  fa[6]=f1.z;  fa[7]=f1.w;
            fa[8]=f2.x;  fa[9]=f2.y;  fa[10]=f2.z; fa[11]=f2.w;
            fa[12]=f3.x; fa[13]=f3.y; fa[14]=f3.z; fa[15]=f3.w;
        }
        v16bf ah, al;
        #pragma unroll
        for (int j = 0; j < 16; ++j) {
            __bf16 h = (__bf16)fa[j];
            ah[j] = h;
            al[j] = (__bf16)(fa[j] - (float)h);
        }

        #pragma unroll
        for (int kk = 0; kk < K_; ++kk) {
            const size_t off =
                ((((size_t)(kk * 4 + s) * 8 + wtile) * 32) + lane) * 16;
            v16bf bh = *(const v16bf*)(wh + off);
            v16bf bl = *(const v16bf*)(wl + off);
            // fp32 ~ hi + lo:  Ah*Bh + Ah*Bl + Al*Bh  (Al*Bl negligible)
            acc[kk] = __builtin_amdgcn_wmma_f32_16x16x32_bf16(
                false, al, false, bh, (short)0, acc[kk], false, false);
            acc[kk] = __builtin_amdgcn_wmma_f32_16x16x32_bf16(
                false, ah, false, bl, (short)0, acc[kk], false, false);
            acc[kk] = __builtin_amdgcn_wmma_f32_16x16x32_bf16(
                false, ah, false, bh, (short)0, acc[kk], false, false);
        }
    }

    // ---- epilogue: tanh gate, tap-weighted sum, store ----
    // C/D layout: lane l, element i -> m = i + 8*(l/16), n = l%16
    const int c = wtile * 16 + (lane & 15);
    #pragma unroll
    for (int i = 0; i < 8; ++i) {
        const int mm = i + 8 * half;
        float o = 0.f;
        #pragma unroll
        for (int kk = 0; kk < K_; ++kk)
            o += lds_x[(mm + kk) * C_ + c] * FAST_TANH(acc[kk][i]);
        const int row = l0 + mm;
        if (row < LP_)
            out[((size_t)bidx * LP_ + row) * C_ + c] = o;
    }
}

extern "C" void kernel_launch(void* const* d_in, const int* in_sizes, int n_in,
                              void* d_out, int out_size, void* d_ws, size_t ws_size,
                              hipStream_t stream) {
    const float* x = (const float*)d_in[0];   // (8, 4096, 128) f32
    const float* w = (const float*)d_in[1];   // (128, 128, 7) f32
    float* out = (float*)d_out;               // (8, 4090, 128) f32

    // scratch: bf16 hi fragments then bf16 lo fragments (458,752 bytes total)
    __bf16* wh = (__bf16*)d_ws;
    __bf16* wl = wh + NFRAG_ELEMS;

    prep_weights<<<(NFRAG_ELEMS + 255) / 256, 256, 0, stream>>>(w, wh, wl);
    sidechain_kernel<<<B_ * TILES_, 256, 0, stream>>>(x, wh, wl, out);
}
